// RankLoss_90950227460178
// MI455X (gfx1250) — 8-bit matrix operand order repaired and checked by reading, compiled, not run on hardware
//
#include <hip/hip_runtime.h>

typedef int v8i __attribute__((ext_vector_type(8)));

#define N_ELEMS 8192
#define N_TILES 512

__global__ __launch_bounds__(256) void rankloss_pairs_kernel(
    const float* __restrict__ score, const int* __restrict__ gidx,
    float* __restrict__ ws) {
  __shared__ float s_lds[N_ELEMS];
  __shared__ unsigned char g_lds[N_ELEMS];
  __shared__ float wred[16];

  const int tid = threadIdx.x;

  {
    const float4* s4 = (const float4*)score;
    float4* d4 = (float4*)s_lds;
#pragma unroll
    for (int k = 0; k < 8; ++k) d4[tid + k * 256] = s4[tid + k * 256];
#pragma unroll
    for (int k = 0; k < 32; ++k) {
      const int i = tid + k * 256;
      g_lds[i] = (unsigned char)gidx[i];
    }
  }
  __syncthreads();

  const int bi   = blockIdx.x;
  const int wave = tid >> 5;
  const int lane = tid & 31;
  const int half = lane >> 4;
  const int l15  = lane & 15;

  v8i A = {};
  {
    const int g    = g_lds[bi * 16 + l15];
    const int vsel = 2 * (g >> 4) + ((g & 7) >> 2);
    const int val  = (((g >> 3) & 1) == half) ? (1 << ((g & 3) * 8)) : 0;
#pragma unroll
    for (int r = 0; r < 8; ++r) A[r] = (vsel == r) ? val : 0;
  }

  float row_s[8];
#pragma unroll
  for (int r = 0; r < 8; ++r) row_s[r] = s_lds[bi * 16 + r + 8 * half];

  float acc = 0.0f;
  float cnt = 0.0f;

  for (int bj = bi + wave; bj < N_TILES; bj += 8) {
    const int jg = bj * 16 + l15;

    v8i B = {};
    {
      const int g    = g_lds[jg];
      const int vsel = 2 * (g >> 4) + ((g & 7) >> 2);
      const int val  = (((g >> 3) & 1) == half) ? (1 << ((g & 3) * 8)) : 0;
#pragma unroll
      for (int r = 0; r < 8; ++r) B[r] = (vsel == r) ? val : 0;
    }

    v8i Cz = {};
    v8i M = __builtin_amdgcn_wmma_i32_16x16x64_iu8(
         false, A,  false, B, Cz,
         false,  false);

    const float sj = s_lds[jg];
#pragma unroll
    for (int r = 0; r < 8; ++r) {
      const int ig = bi * 16 + r + 8 * half;
      const bool valid = (M[r] != 0) && (jg > ig);
      const float x = sj - row_s[r];
      const float l = fmaxf(x, 0.0f) + __logf(1.0f + __expf(-fabsf(x)));
      acc += valid ? l : 0.0f;
      cnt += valid ? 1.0f : 0.0f;
    }
  }

#pragma unroll
  for (int off = 16; off > 0; off >>= 1) {
    acc += __shfl_down(acc, off, 32);
    cnt += __shfl_down(cnt, off, 32);
  }
  if (lane == 0) {
    wred[wave * 2]     = acc;
    wred[wave * 2 + 1] = cnt;
  }
  __syncthreads();
  if (tid == 0) {
    float ta = 0.0f, tc = 0.0f;
#pragma unroll
    for (int w = 0; w < 8; ++w) {
      ta += wred[w * 2];
      tc += wred[w * 2 + 1];
    }
    atomicAdd(&ws[0], ta);
    atomicAdd(&ws[1], tc);
  }
}

__global__ void rankloss_finalize_kernel(const float* __restrict__ ws,
                                         float* __restrict__ out) {
  out[0] = ws[0] / ws[1];
}

extern "C" void kernel_launch(void* const* d_in, const int* in_sizes, int n_in,
                              void* d_out, int out_size, void* d_ws, size_t ws_size,
                              hipStream_t stream) {
  const float* score = (const float*)d_in[0];
  const int* gidx    = (const int*)d_in[1];
  float* ws          = (float*)d_ws;
  float* out         = (float*)d_out;

  hipMemsetAsync(ws, 0, 2 * sizeof(float), stream);
  rankloss_pairs_kernel<<<N_TILES, 256, 0, stream>>>(score, gidx, ws);
  rankloss_finalize_kernel<<<1, 1, 0, stream>>>(ws, out);
}
